// Mamba3Core_43250320670948
// MI455X (gfx1250) — matrix operand order repaired and checked by reading, not compiled, not run on hardware
//
#include <hip/hip_runtime.h>
#include <hip/hip_bf16.h>

#define D_DIM   2048
#define NC_DIM  2
#define DS_DIM  16
#define BL_DIM  1024            // B*L = 2*512
#define N1_DIM  4096            // NC*D

typedef __bf16 bf16;
typedef __attribute__((ext_vector_type(16))) __bf16 v16bf;
typedef __attribute__((ext_vector_type(8)))  __bf16 v8bf;
typedef __attribute__((ext_vector_type(8)))  float  v8f;
typedef __attribute__((ext_vector_type(4)))  float  v4f;
typedef __attribute__((ext_vector_type(4)))  int    v4i;

typedef __attribute__((address_space(1))) v4i* gptr_v4i;   // global
typedef __attribute__((address_space(3))) v4i* lptr_v4i;   // LDS

#if defined(__has_builtin)
#if __has_builtin(__builtin_amdgcn_global_load_async_to_lds_b128)
#define M3_ASYNC_LDS 1
#endif
#endif

#ifdef M3_ASYNC_LDS
__device__ __forceinline__ void m3_async_b128(const bf16* g, bf16* l)
{
    __builtin_amdgcn_global_load_async_to_lds_b128(
        (gptr_v4i)(g), (lptr_v4i)(l), 0, 0);
}
__device__ __forceinline__ void m3_wait_async0()
{
#if __has_builtin(__builtin_amdgcn_s_wait_asynccnt)
    __builtin_amdgcn_s_wait_asynccnt(0);
#else
    asm volatile("s_wait_asynccnt 0x0" ::: "memory");
#endif
}
#endif

// ---------------------------------------------------------------------------
// fp32 -> bf16 bulk conversion (one-time; keeps GEMM hot loop math-free)
// ---------------------------------------------------------------------------
__global__ __launch_bounds__(256)
void m3_f32_to_bf16_kernel(const float* __restrict__ s, bf16* __restrict__ d)
{
    const int i = (blockIdx.x * 256 + threadIdx.x) * 8;
    v4f a = *reinterpret_cast<const v4f*>(s + i);
    v4f b = *reinterpret_cast<const v4f*>(s + i + 4);
    v8bf o;
#pragma unroll
    for (int e = 0; e < 4; ++e) { o[e] = (bf16)a[e]; o[4 + e] = (bf16)b[e]; }
    *reinterpret_cast<v8bf*>(d + i) = o;
}

// ---------------------------------------------------------------------------
// Kscale[c*D+d] = sum_s ( C_real*B_real - C_imag*B_imag )
// ---------------------------------------------------------------------------
__global__ __launch_bounds__(256)
void m3_kscale_kernel(const float* __restrict__ Br, const float* __restrict__ Bi,
                      const float* __restrict__ Cr, const float* __restrict__ Ci,
                      float* __restrict__ Ks)
{
    int t = blockIdx.x * 256 + threadIdx.x;        // < NC*D = 4096
    int base = t * DS_DIM;
    float acc = 0.f;
#pragma unroll
    for (int s = 0; s < DS_DIM; ++s)
        acc += Cr[base + s] * Br[base + s] - Ci[base + s] * Bi[base + s];
    Ks[t] = acc;
}

// ---------------------------------------------------------------------------
// C[M,N] = A[M,K] @ B[N,K]^T  -- bf16 operands in memory, fp32 accum.
// Block tile 128x128, K tile 32; 8 waves (2 M x 4 N), wave tile 64x32.
// Double-buffered LDS; staging via async global->LDS when available.
// ---------------------------------------------------------------------------
#define BM   128
#define BN   128
#define BK   32
#define LDSK 48   // 96B row stride: keeps all 16B ds_load_b128 aligned

__global__ __launch_bounds__(256)
void m3_gemm_bf16_wmma(const bf16* __restrict__ A, const bf16* __restrict__ Bw,
                       float* __restrict__ C, int M, int N, int K)
{
    __shared__ bf16 As[2][BM][LDSK];
    __shared__ bf16 Bs[2][BN][LDSK];

    const int tid  = threadIdx.x;
    const int lane = tid & 31;
    const int wid  = tid >> 5;       // 0..7
    const int wm   = wid >> 2;       // 0..1  (M direction)
    const int wn   = wid & 3;        // 0..3  (N direction)
    const int m0   = blockIdx.y * BM;
    const int n0   = blockIdx.x * BN;

    // loader: each thread moves 16 bf16 (32B) of A and of B per K tile
    const int lrow = tid >> 1;           // 0..127
    const int lseg = (tid & 1) * 16;     // 0 or 16 (bf16 elements)

    const bf16* Aldp = A  + (size_t)(m0 + lrow) * K + lseg;
    const bf16* Bldp = Bw + (size_t)(n0 + lrow) * K + lseg;

    v8f acc[4][2];
#pragma unroll
    for (int i = 0; i < 4; ++i)
#pragma unroll
        for (int j = 0; j < 2; ++j)
#pragma unroll
            for (int e = 0; e < 8; ++e) acc[i][j][e] = 0.0f;

#ifndef M3_ASYNC_LDS
    v8bf sa0, sa1, sb0, sb1;             // register staging (fallback path)
#endif

    // ---- prologue: issue tile 0 into buffer 0 ----
#ifdef M3_ASYNC_LDS
    m3_async_b128(Aldp,     &As[0][lrow][lseg]);
    m3_async_b128(Aldp + 8, &As[0][lrow][lseg + 8]);
    m3_async_b128(Bldp,     &Bs[0][lrow][lseg]);
    m3_async_b128(Bldp + 8, &Bs[0][lrow][lseg + 8]);
#else
    sa0 = *reinterpret_cast<const v8bf*>(Aldp);
    sa1 = *reinterpret_cast<const v8bf*>(Aldp + 8);
    sb0 = *reinterpret_cast<const v8bf*>(Bldp);
    sb1 = *reinterpret_cast<const v8bf*>(Bldp + 8);
#endif

    const int kTiles = K / BK;
    for (int kt = 0; kt < kTiles; ++kt) {
        const int cur = kt & 1;

        // ---- finish staging of buffer `cur` ----
#ifdef M3_ASYNC_LDS
        m3_wait_async0();
#else
        *reinterpret_cast<v8bf*>(&As[cur][lrow][lseg])     = sa0;
        *reinterpret_cast<v8bf*>(&As[cur][lrow][lseg + 8]) = sa1;
        *reinterpret_cast<v8bf*>(&Bs[cur][lrow][lseg])     = sb0;
        *reinterpret_cast<v8bf*>(&Bs[cur][lrow][lseg + 8]) = sb1;
#endif
        __syncthreads();

        // ---- issue staging of tile kt+1 into buffer cur^1 ----
        if (kt + 1 < kTiles) {
            const bf16* ap = Aldp + (size_t)(kt + 1) * BK;
            const bf16* bp = Bldp + (size_t)(kt + 1) * BK;
#ifdef M3_ASYNC_LDS
            m3_async_b128(ap,     &As[cur ^ 1][lrow][lseg]);
            m3_async_b128(ap + 8, &As[cur ^ 1][lrow][lseg + 8]);
            m3_async_b128(bp,     &Bs[cur ^ 1][lrow][lseg]);
            m3_async_b128(bp + 8, &Bs[cur ^ 1][lrow][lseg + 8]);
#else
            sa0 = *reinterpret_cast<const v8bf*>(ap);
            sa1 = *reinterpret_cast<const v8bf*>(ap + 8);
            sb0 = *reinterpret_cast<const v8bf*>(bp);
            sb1 = *reinterpret_cast<const v8bf*>(bp + 8);
#endif
            // pull tile kt+2 toward L2 while tile kt computes
            __builtin_prefetch(ap + BK, 0, 0);
            __builtin_prefetch(bp + BK, 0, 0);
        }

        // ---- fragments per documented CDNA5 16-bit VGPR layouts ----
        const int half = lane >> 4;      // 0: lanes 0-15, 1: lanes 16-31
        const int lr   = lane & 15;
        v16bf afrag[4], bfrag[2];
        {
            const int akb = half * 8;    // A: K 0-7/16-23 vs 8-15/24-31
#pragma unroll
            for (int fm = 0; fm < 4; ++fm) {
                const int row = wm * 64 + fm * 16 + lr;
                v8bf lo = *reinterpret_cast<const v8bf*>(&As[cur][row][akb]);
                v8bf hi = *reinterpret_cast<const v8bf*>(&As[cur][row][akb + 16]);
#pragma unroll
                for (int e = 0; e < 8; ++e) { afrag[fm][e] = lo[e]; afrag[fm][8 + e] = hi[e]; }
            }
            const int bkb = half * 8;    // B: K 0-7/16-23 vs 8-15/24-31, lane = N
#pragma unroll
            for (int fn = 0; fn < 2; ++fn) {
                const int col = wn * 32 + fn * 16 + lr;
                v8bf lo = *reinterpret_cast<const v8bf*>(&Bs[cur][col][bkb]);
                v8bf hi = *reinterpret_cast<const v8bf*>(&Bs[cur][col][bkb + 16]);
#pragma unroll
                for (int e = 0; e < 8; ++e) { bfrag[fn][e] = lo[e]; bfrag[fn][8 + e] = hi[e]; }
            }
        }

#pragma unroll
        for (int fm = 0; fm < 4; ++fm)
#pragma unroll
            for (int fn = 0; fn < 2; ++fn)
                acc[fm][fn] = __builtin_amdgcn_wmma_f32_16x16x32_bf16(
                    false, afrag[fm], false, bfrag[fn],
                    (short)0, acc[fm][fn], false, false);
    }

    // ---- epilogue: C/D layout -> VGPR r holds M=r (lanes 0-15) / M=8+r ----
    const int lr   = lane & 15;
    const int rofs = (lane >> 4) * 8;
#pragma unroll
    for (int fm = 0; fm < 4; ++fm)
#pragma unroll
        for (int fn = 0; fn < 2; ++fn)
#pragma unroll
            for (int r = 0; r < 8; ++r) {
                const int grow = m0 + wm * 64 + fm * 16 + rofs + r;
                const int gcol = n0 + wn * 32 + fn * 16 + lr;
                C[(size_t)grow * N + gcol] = acc[fm][fn][r];
            }
}

// ---------------------------------------------------------------------------
// nr = B_real*xi, ni = B_imag*xi (536 MB non-temporal stream)
// fused: y_bf16 = bf16(xi * Kscale)  (GEMM2's pre-scaled A operand)
// ---------------------------------------------------------------------------
__global__ __launch_bounds__(256)
void m3_nrni_kernel(const float* __restrict__ xproj,
                    const float* __restrict__ Br, const float* __restrict__ Bi,
                    const float* __restrict__ Ks,
                    float* __restrict__ nr, float* __restrict__ ni,
                    bf16* __restrict__ ybf)
{
    const int idx = blockIdx.x * 256 + threadIdx.x;   // < BL*NC*D = 4194304
    const float xi = xproj[idx];
    const int cd = idx & (N1_DIM - 1);
    ybf[idx] = (bf16)(xi * Ks[cd]);

    const size_t obase = (size_t)idx * DS_DIM;
    const v4f* br = reinterpret_cast<const v4f*>(Br + cd * DS_DIM);
    const v4f* bi = reinterpret_cast<const v4f*>(Bi + cd * DS_DIM);
#pragma unroll
    for (int j = 0; j < 4; ++j) {
        v4f b = br[j], o;
#pragma unroll
        for (int e = 0; e < 4; ++e) o[e] = b[e] * xi;
        __builtin_nontemporal_store(o, reinterpret_cast<v4f*>(nr + obase) + j);
    }
#pragma unroll
    for (int j = 0; j < 4; ++j) {
        v4f b = bi[j], o;
#pragma unroll
        for (int e = 0; e < 4; ++e) o[e] = b[e] * xi;
        __builtin_nontemporal_store(o, reinterpret_cast<v4f*>(ni + obase) + j);
    }
}

// ---------------------------------------------------------------------------
// out1 = x + rmsnorm(out_pre) * norm_w    (one block per row)
// ---------------------------------------------------------------------------
__global__ __launch_bounds__(256)
void m3_residual_rmsnorm_kernel(const float* __restrict__ x,
                                const float* __restrict__ pre,
                                const float* __restrict__ w,
                                float* __restrict__ out)
{
    __shared__ float red[256];
    const int row = blockIdx.x;
    const int tid = threadIdx.x;
    const float* p = pre + (size_t)row * D_DIM;

    float v[8];
    float ss = 0.f;
#pragma unroll
    for (int j = 0; j < 8; ++j) {
        v[j] = p[tid + j * 256];
        ss += v[j] * v[j];
    }
    red[tid] = ss;
    __syncthreads();
    for (int s = 128; s > 0; s >>= 1) {
        if (tid < s) red[tid] += red[tid + s];
        __syncthreads();
    }
    const float scale = rsqrtf(red[0] / (float)D_DIM + 1.1920928955078125e-07f);
#pragma unroll
    for (int j = 0; j < 8; ++j) {
        const int d = tid + j * 256;
        out[(size_t)row * D_DIM + d] = x[(size_t)row * D_DIM + d] + v[j] * scale * w[d];
    }
}

// ---------------------------------------------------------------------------
extern "C" void kernel_launch(void* const* d_in, const int* in_sizes, int n_in,
                              void* d_out, int out_size, void* d_ws, size_t ws_size,
                              hipStream_t stream)
{
    const float* x      = (const float*)d_in[0];  // [2,512,2048]
    const float* W_in   = (const float*)d_in[1];  // [4096,2048]
    const float* W_out  = (const float*)d_in[2];  // [2048,4096]
    // d_in[3] = A_theta: dead (multiplies zero initial state)
    const float* B_real = (const float*)d_in[4];  // [2,2048,16]
    const float* B_imag = (const float*)d_in[5];
    const float* C_real = (const float*)d_in[6];
    const float* C_imag = (const float*)d_in[7];
    const float* norm_w = (const float*)d_in[8];  // [2048]

    float* out  = (float*)d_out;
    float* out1 = out;                                   // 2*512*2048
    float* nr   = out1 + (size_t)BL_DIM * D_DIM;         // 2*512*2*2048*16
    float* ni   = nr   + (size_t)BL_DIM * N1_DIM * DS_DIM;

    // workspace carve-up (fp32 regions first for alignment)
    float* xproj = (float*)d_ws;                          // [1024,4096] f32
    float* pre   = xproj + (size_t)BL_DIM * N1_DIM;       // [1024,2048] f32
    float* Ks    = pre   + (size_t)BL_DIM * D_DIM;        // [4096]      f32
    bf16*  xbf   = (bf16*)(Ks + N1_DIM);                  // [1024,2048] bf16
    bf16*  wibf  = xbf  + (size_t)BL_DIM * D_DIM;         // [4096,2048] bf16
    bf16*  wobf  = wibf + (size_t)N1_DIM * D_DIM;         // [2048,4096] bf16
    bf16*  ybf   = wobf + (size_t)D_DIM * N1_DIM;         // [1024,4096] bf16

    // one-time bf16 conversions
    m3_f32_to_bf16_kernel<<<(BL_DIM * D_DIM) / (256 * 8), 256, 0, stream>>>(x, xbf);
    m3_f32_to_bf16_kernel<<<(N1_DIM * D_DIM) / (256 * 8), 256, 0, stream>>>(W_in, wibf);
    m3_f32_to_bf16_kernel<<<(D_DIM * N1_DIM) / (256 * 8), 256, 0, stream>>>(W_out, wobf);

    m3_kscale_kernel<<<N1_DIM / 256, 256, 0, stream>>>(B_real, B_imag, C_real, C_imag, Ks);

    // GEMM1: xproj = x @ W_in^T   (M=1024, N=4096, K=2048)
    m3_gemm_bf16_wmma<<<dim3(N1_DIM / BN, BL_DIM / BM), 256, 0, stream>>>(
        xbf, wibf, xproj, BL_DIM, N1_DIM, D_DIM);

    // nr / ni streaming outputs + pre-scaled bf16 y for GEMM2
    m3_nrni_kernel<<<(BL_DIM * N1_DIM) / 256, 256, 0, stream>>>(
        xproj, B_real, B_imag, Ks, nr, ni, ybf);

    // GEMM2: pre = y @ W_out^T   (M=1024, N=2048, K=4096)
    m3_gemm_bf16_wmma<<<dim3(D_DIM / BN, BL_DIM / BM), 256, 0, stream>>>(
        ybf, wobf, pre, BL_DIM, D_DIM, N1_DIM);

    // residual + rmsnorm
    m3_residual_rmsnorm_kernel<<<BL_DIM, 256, 0, stream>>>(x, pre, norm_w, out1);
}
